// EGCL_68195490726193
// MI455X (gfx1250) — matrix operand order repaired and checked by reading, not compiled, not run on hardware
//
#include <hip/hip_runtime.h>
#include <hip/hip_bf16.h>
#include <math.h>

// ---------------------------------------------------------------------------
// EGNN layer (EGCL) for MI455X / gfx1250, wave32 + WMMA bf16.
//   edge:  m_ij = silu(silu([h_r|h_c|rad] @ We1 + be1) @ We2 + be2)
//          w    = silu(m_ij @ Wc1 + bc1) @ Wc2 + bc2
//          coord_out += scatter_r(diffn * w);  m_i += scatter_r(m_ij)
//   node:  h_out = h + silu([h|m_i] @ Wn1 + bn1) @ Wn2 + bn2
// All GEMMs run on v_wmma_f32_16x16x32_bf16 (16 rows per wave, 8 N-tiles).
// A-fragments are hoisted into registers across the N-tile loop; B-fragments
// stream from a pre-swizzled bf16 weight buffer (L2-resident).
// ---------------------------------------------------------------------------

typedef __bf16 bf16_t;
typedef __attribute__((ext_vector_type(16))) __bf16 v16bf;
typedef __attribute__((ext_vector_type(8)))  __bf16 v8bf;
typedef __attribute__((ext_vector_type(4)))  __bf16 v4bf;
typedef __attribute__((ext_vector_type(8)))  float  v8f;

#define DIMF 128
#define A_STRIDE 264   // 256 cols + 8 pad (bf16 elems); row = 528 B (16B mult)
#define B_STRIDE 136   // 128 cols + 8 pad; row = 272 B (16B mult)
#define WAVES_PER_BLK 4

struct __align__(16) WaveBuf {
    bf16_t A[16 * A_STRIDE];   // 8448 B : gathered input tile / m_ij tile
    bf16_t B[16 * B_STRIDE];   // 4352 B : m1 tile / w1 tile
    int    ri[16];
    int    ci[16];
    float  rad[16];
    float  wgt[16];
};                             // 13056 B per wave; x4 = 52224 B static LDS

// silu with hardware rcp approximation (avoids IEEE div_scale/div_fmas chain)
__device__ __forceinline__ float silu_f(float x) {
    return x * __builtin_amdgcn_rcpf(1.0f + __expf(-x));
}

// A fragment (16x32 bf16), ISA layout: lanes 0-15 row M=lane hold K 0-7 & 16-23,
// lanes 16-31 row M=lane-16 hold K 8-15 & 24-31. Two 16B LDS loads per lane.
__device__ __forceinline__ v16bf load_a_frag(const bf16_t* Alds, int stride,
                                             int kt, int lane) {
    const int row = lane & 15;
    const int kb  = (lane >> 4) << 3;              // 0 or 8
    const bf16_t* p = Alds + row * stride + kt * 32 + kb;
    v8bf lo = *(const v8bf*)(p);
    v8bf hi = *(const v8bf*)(p + 16);
    v16bf a;
#pragma unroll
    for (int i = 0; i < 8; ++i) { a[i] = lo[i]; a[i + 8] = hi[i]; }
    return a;
}

// B fragments are pre-swizzled by the pack kernel into per-lane contiguous
// 32B chunks: frag f, lane l -> 16 bf16 at p + (f*32 + l)*16.
__device__ __forceinline__ v16bf load_b_frag(const bf16_t* p, int frag, int lane) {
    return *(const v16bf*)(p + ((size_t)frag * 32 + lane) * 16);
}

__device__ __forceinline__ v8f wmma_bf16(v16bf a, v16bf b, v8f c) {
    return __builtin_amdgcn_wmma_f32_16x16x32_bf16(
        /*neg_a=*/false, a, /*neg_b=*/false, b,
        /*c_mod=*/(short)0, c, /*reuse_a=*/false, /*reuse_b=*/false);
}

// ---------------------------------------------------------------------------
// Weight pack: W[K x ncols] f32 -> bf16 fragments in WMMA B layout.
// B (32x16) layout: lanes 0-15 col N=lane hold K rows 0-7 & 16-23; lanes 16-31 hold
// K rows 8-15 & 24-31; element j is K row (lane>=16?8:0)+(j<8?j:j+8).
// c2flag: pad a [K x 1] vector into N=16 tiles with only col 0 live.
// ---------------------------------------------------------------------------
__global__ void egcl_pack_kernel(const float* __restrict__ W, bf16_t* __restrict__ dst,
                                 int KT, int NT, int ncols, int kmax, int c2flag) {
    int t = blockIdx.x * blockDim.x + threadIdx.x;
    int nfrag = KT * NT;
    if (t >= nfrag * 32) return;
    int f = t >> 5, lane = t & 31;
    int kt = f / NT, nt = f % NT;
    int kb  = (lane >> 4) << 3;                    // 0 or 8
    int col = nt * 16 + (lane & 15);
#pragma unroll
    for (int j = 0; j < 16; ++j) {
        int k = kt * 32 + kb + (j < 8 ? j : j + 8);
        float v = 0.0f;
        if (k < kmax) {
            if (c2flag) v = (col == 0) ? W[k] : 0.0f;
            else        v = W[(size_t)k * ncols + col];
        }
        dst[(size_t)t * 16 + j] = (bf16_t)v;
    }
}

__global__ void egcl_init_kernel(const float* __restrict__ coord,
                                 const float* __restrict__ We1,
                                 float* __restrict__ m_i,
                                 float* __restrict__ coord_out,
                                 float* __restrict__ w257, int N_) {
    int i = blockIdx.x * blockDim.x + threadIdx.x;
    if (i < N_ * DIMF) m_i[i] = 0.0f;
    if (i < N_ * 3)    coord_out[i] = coord[i];
    if (i < DIMF)      w257[i] = We1[(size_t)256 * DIMF + i];   // rad row of We1
}

// ---------------------------------------------------------------------------
// Edge kernel: one wave = 16 edges; full 128-wide MLP chain in registers/LDS.
// ---------------------------------------------------------------------------
__global__ __launch_bounds__(128, 1)
void egcl_edge_kernel(const float* __restrict__ h, const float* __restrict__ coord,
                      const int* __restrict__ ei,
                      const bf16_t* __restrict__ pWe1, const float* __restrict__ be1,
                      const bf16_t* __restrict__ pWe2, const float* __restrict__ be2,
                      const bf16_t* __restrict__ pWc1, const float* __restrict__ bc1,
                      const bf16_t* __restrict__ pWc2, const float* __restrict__ bc2,
                      const float* __restrict__ w257,
                      float* __restrict__ m_i, float* __restrict__ coord_out, int E_) {
    __shared__ WaveBuf sbuf[WAVES_PER_BLK];
    const int lane  = threadIdx.x & 31;
    const int wave  = threadIdx.x >> 5;
    WaveBuf& S      = sbuf[wave];
    const int tile  = blockIdx.x * WAVES_PER_BLK + wave;
    const int ebase = tile * 16;
    const int lcol  = lane & 15;
    const int rbase = (lane >> 4) << 3;

    // ---- edge indices, diff, rad (lanes 0-15, one edge each) ----
    float dxn = 0.f, dyn = 0.f, dzn = 0.f;
    if (lane < 16) {
        int e = ebase + lane; if (e >= E_) e = E_ - 1;
        int r = ei[e];
        int c = ei[E_ + e];
        S.ri[lane] = r; S.ci[lane] = c;
        float dx = coord[r * 3 + 0] - coord[c * 3 + 0];
        float dy = coord[r * 3 + 1] - coord[c * 3 + 1];
        float dz = coord[r * 3 + 2] - coord[c * 3 + 2];
        float rad = dx * dx + dy * dy + dz * dz;
        S.rad[lane] = rad;
        float inv = __builtin_amdgcn_rcpf(sqrtf(rad) + 1e-8f);
        dxn = dx * inv; dyn = dy * inv; dzn = dz * inv;
    }
    __syncthreads();

    // ---- gather A = [h_r | h_c] as bf16, row-major, stride A_STRIDE ----
    for (int m = 0; m < 16; ++m) {
        int r = S.ri[m], c = S.ci[m];
        float4 v0 = *(const float4*)(h + (size_t)r * DIMF + lane * 4);
        float4 v1 = *(const float4*)(h + (size_t)c * DIMF + lane * 4);
        v4bf t0, t1;
        t0[0] = (bf16_t)v0.x; t0[1] = (bf16_t)v0.y; t0[2] = (bf16_t)v0.z; t0[3] = (bf16_t)v0.w;
        t1[0] = (bf16_t)v1.x; t1[1] = (bf16_t)v1.y; t1[2] = (bf16_t)v1.z; t1[3] = (bf16_t)v1.w;
        *(v4bf*)(&S.A[m * A_STRIDE + lane * 4])        = t0;
        *(v4bf*)(&S.A[m * A_STRIDE + 128 + lane * 4])  = t1;
    }
    __syncthreads();

    // ---- Layer 1: m1 = silu(A @ We1[0:256] + rad*We1[256] + be1) -> S.B ----
    {
        v16bf af[8];
#pragma unroll
        for (int kt = 0; kt < 8; ++kt) af[kt] = load_a_frag(S.A, A_STRIDE, kt, lane);
        for (int nt = 0; nt < 8; ++nt) {
            float bv = be1[nt * 16 + lcol];
            v8f acc;
#pragma unroll
            for (int i = 0; i < 8; ++i) acc[i] = bv;
#pragma unroll
            for (int kt = 0; kt < 8; ++kt)
                acc = wmma_bf16(af[kt], load_b_frag(pWe1, kt * 8 + nt, lane), acc);
            float wr = w257[nt * 16 + lcol];
#pragma unroll
            for (int i = 0; i < 8; ++i) {
                float x = silu_f(acc[i] + S.rad[rbase + i] * wr);
                S.B[(rbase + i) * B_STRIDE + nt * 16 + lcol] = (bf16_t)x;
            }
        }
    }
    __syncthreads();

    // ---- Layer 2: m_ij = silu(m1 @ We2 + be2) -> S.A ; scatter into m_i ----
    {
        v16bf af[4];
#pragma unroll
        for (int kt = 0; kt < 4; ++kt) af[kt] = load_a_frag(S.B, B_STRIDE, kt, lane);
        for (int nt = 0; nt < 8; ++nt) {
            float bv = be2[nt * 16 + lcol];
            v8f acc;
#pragma unroll
            for (int i = 0; i < 8; ++i) acc[i] = bv;
#pragma unroll
            for (int kt = 0; kt < 4; ++kt)
                acc = wmma_bf16(af[kt], load_b_frag(pWe2, kt * 8 + nt, lane), acc);
#pragma unroll
            for (int i = 0; i < 8; ++i) {
                int row = rbase + i;
                float x = silu_f(acc[i]);
                S.A[row * B_STRIDE + nt * 16 + lcol] = (bf16_t)x;
                if (ebase + row < E_)
                    atomicAdd(&m_i[(size_t)S.ri[row] * DIMF + nt * 16 + lcol], x);
            }
        }
    }
    __syncthreads();

    // ---- Coord layer 1: w1 = silu(m_ij @ Wc1 + bc1) -> S.B ----
    {
        v16bf af[4];
#pragma unroll
        for (int kt = 0; kt < 4; ++kt) af[kt] = load_a_frag(S.A, B_STRIDE, kt, lane);
        for (int nt = 0; nt < 8; ++nt) {
            float bv = bc1[nt * 16 + lcol];
            v8f acc;
#pragma unroll
            for (int i = 0; i < 8; ++i) acc[i] = bv;
#pragma unroll
            for (int kt = 0; kt < 4; ++kt)
                acc = wmma_bf16(af[kt], load_b_frag(pWc1, kt * 8 + nt, lane), acc);
#pragma unroll
            for (int i = 0; i < 8; ++i) {
                float x = silu_f(acc[i]);
                S.B[(rbase + i) * B_STRIDE + nt * 16 + lcol] = (bf16_t)x;
            }
        }
    }
    __syncthreads();

    // ---- Coord layer 2: weight = w1 @ Wc2 + bc2 (N padded to 16, col 0) ----
    {
        v8f acc;
        float bv = (lcol == 0) ? bc2[0] : 0.0f;
#pragma unroll
        for (int i = 0; i < 8; ++i) acc[i] = bv;
#pragma unroll
        for (int kt = 0; kt < 4; ++kt)
            acc = wmma_bf16(load_a_frag(S.B, B_STRIDE, kt, lane),
                            load_b_frag(pWc2, kt, lane), acc);
        if (lcol == 0) {
#pragma unroll
            for (int i = 0; i < 8; ++i) S.wgt[rbase + i] = acc[i];
        }
    }
    __syncthreads();

    // ---- coord scatter: coord_out[r] += diffn * weight ----
    if (lane < 16 && (ebase + lane) < E_) {
        float w = S.wgt[lane];
        int r = S.ri[lane];
        atomicAdd(&coord_out[r * 3 + 0], dxn * w);
        atomicAdd(&coord_out[r * 3 + 1], dyn * w);
        atomicAdd(&coord_out[r * 3 + 2], dzn * w);
    }
}

// ---------------------------------------------------------------------------
// Node kernel: one wave = 16 nodes; h_out = h + Lin(silu(Lin([h|m_i])))
// ---------------------------------------------------------------------------
__global__ __launch_bounds__(128, 1)
void egcl_node_kernel(const float* __restrict__ h, const float* __restrict__ m_i,
                      const bf16_t* __restrict__ pWn1, const float* __restrict__ bn1,
                      const bf16_t* __restrict__ pWn2, const float* __restrict__ bn2,
                      float* __restrict__ h_out, int N_) {
    __shared__ WaveBuf sbuf[WAVES_PER_BLK];
    const int lane  = threadIdx.x & 31;
    const int wave  = threadIdx.x >> 5;
    WaveBuf& S      = sbuf[wave];
    const int tile  = blockIdx.x * WAVES_PER_BLK + wave;
    const int nbase = tile * 16;
    const int lcol  = lane & 15;
    const int rbase = (lane >> 4) << 3;

    // ---- gather A = [h | m_i] bf16 ----
    for (int m = 0; m < 16; ++m) {
        int node = nbase + m; if (node >= N_) node = N_ - 1;
        float4 v0 = *(const float4*)(h   + (size_t)node * DIMF + lane * 4);
        float4 v1 = *(const float4*)(m_i + (size_t)node * DIMF + lane * 4);
        v4bf t0, t1;
        t0[0] = (bf16_t)v0.x; t0[1] = (bf16_t)v0.y; t0[2] = (bf16_t)v0.z; t0[3] = (bf16_t)v0.w;
        t1[0] = (bf16_t)v1.x; t1[1] = (bf16_t)v1.y; t1[2] = (bf16_t)v1.z; t1[3] = (bf16_t)v1.w;
        *(v4bf*)(&S.A[m * A_STRIDE + lane * 4])       = t0;
        *(v4bf*)(&S.A[m * A_STRIDE + 128 + lane * 4]) = t1;
    }
    __syncthreads();

    // ---- Layer 1: silu([h|m_i] @ Wn1 + bn1) -> S.B ----
    {
        v16bf af[8];
#pragma unroll
        for (int kt = 0; kt < 8; ++kt) af[kt] = load_a_frag(S.A, A_STRIDE, kt, lane);
        for (int nt = 0; nt < 8; ++nt) {
            float bv = bn1[nt * 16 + lcol];
            v8f acc;
#pragma unroll
            for (int i = 0; i < 8; ++i) acc[i] = bv;
#pragma unroll
            for (int kt = 0; kt < 8; ++kt)
                acc = wmma_bf16(af[kt], load_b_frag(pWn1, kt * 8 + nt, lane), acc);
#pragma unroll
            for (int i = 0; i < 8; ++i) {
                float x = silu_f(acc[i]);
                S.B[(rbase + i) * B_STRIDE + nt * 16 + lcol] = (bf16_t)x;
            }
        }
    }
    __syncthreads();

    // ---- Layer 2 + residual: h_out = h + (. @ Wn2 + bn2) ----
    {
        v16bf af[4];
#pragma unroll
        for (int kt = 0; kt < 4; ++kt) af[kt] = load_a_frag(S.B, B_STRIDE, kt, lane);
        for (int nt = 0; nt < 8; ++nt) {
            float bv = bn2[nt * 16 + lcol];
            v8f acc;
#pragma unroll
            for (int i = 0; i < 8; ++i) acc[i] = bv;
#pragma unroll
            for (int kt = 0; kt < 4; ++kt)
                acc = wmma_bf16(af[kt], load_b_frag(pWn2, kt * 8 + nt, lane), acc);
#pragma unroll
            for (int i = 0; i < 8; ++i) {
                int row  = rbase + i;
                int node = nbase + row;
                if (node < N_) {
                    size_t idx = (size_t)node * DIMF + nt * 16 + lcol;
                    h_out[idx] = h[idx] + acc[i];
                }
            }
        }
    }
}

// ---------------------------------------------------------------------------
extern "C" void kernel_launch(void* const* d_in, const int* in_sizes, int n_in,
                              void* d_out, int out_size, void* d_ws, size_t ws_size,
                              hipStream_t stream) {
    const float* h     = (const float*)d_in[0];
    const float* coord = (const float*)d_in[1];
    const int*   ei    = (const int*)d_in[2];
    const float* We1 = (const float*)d_in[3];  const float* be1 = (const float*)d_in[4];
    const float* We2 = (const float*)d_in[5];  const float* be2 = (const float*)d_in[6];
    const float* Wc1 = (const float*)d_in[7];  const float* bc1 = (const float*)d_in[8];
    const float* Wc2 = (const float*)d_in[9];  const float* bc2 = (const float*)d_in[10];
    const float* Wn1 = (const float*)d_in[11]; const float* bn1 = (const float*)d_in[12];
    const float* Wn2 = (const float*)d_in[13]; const float* bn2 = (const float*)d_in[14];

    const int N_ = in_sizes[0] / DIMF;
    const int E_ = in_sizes[2] / 2;

    float* h_out     = (float*)d_out;
    float* coord_out = (float*)d_out + (size_t)N_ * DIMF;

    // workspace layout (256B aligned chunks)
    char*  ws  = (char*)d_ws;
    size_t off = 0;
    auto alloc = [&](size_t bytes) -> char* {
        char* p = ws + off;
        off = (off + bytes + 255) & ~(size_t)255;
        return p;
    };
    float*  m_i  = (float*)alloc((size_t)N_ * DIMF * sizeof(float));
    bf16_t* pWe1 = (bf16_t*)alloc(8 * 8 * 1024);   // 64 frags * 1KB
    bf16_t* pWe2 = (bf16_t*)alloc(4 * 8 * 1024);
    bf16_t* pWc1 = (bf16_t*)alloc(4 * 8 * 1024);
    bf16_t* pWc2 = (bf16_t*)alloc(4 * 1 * 1024);
    bf16_t* pWn1 = (bf16_t*)alloc(8 * 8 * 1024);
    bf16_t* pWn2 = (bf16_t*)alloc(4 * 8 * 1024);
    float*  w257 = (float*)alloc(DIMF * sizeof(float));

    // init: zero m_i, seed coord_out with coord, extract We1's rad row
    {
        int total = N_ * DIMF;
        egcl_init_kernel<<<(total + 255) / 256, 256, 0, stream>>>(
            coord, We1, m_i, coord_out, w257, N_);
    }
    // pack weights into WMMA-B fragment order (bf16)
    auto pack = [&](const float* W, bf16_t* dst, int KT, int NT, int ncols,
                    int kmax, int c2) {
        int threads = KT * NT * 32;
        egcl_pack_kernel<<<(threads + 255) / 256, 256, 0, stream>>>(
            W, dst, KT, NT, ncols, kmax, c2);
    };
    pack(We1, pWe1, 8, 8, DIMF, 256, 0);   // rows 0..255 (row 256 -> w257)
    pack(We2, pWe2, 4, 8, DIMF, 128, 0);
    pack(Wc1, pWc1, 4, 8, DIMF, 128, 0);
    pack(Wc2, pWc2, 4, 1, 1,    128, 1);   // [128x1] padded to col 0 of N-tile
    pack(Wn1, pWn1, 8, 8, DIMF, 256, 0);
    pack(Wn2, pWn2, 4, 8, DIMF, 128, 0);

    // edge phase
    {
        int tiles  = (E_ + 15) / 16;
        int blocks = (tiles + WAVES_PER_BLK - 1) / WAVES_PER_BLK;
        egcl_edge_kernel<<<blocks, 128, 0, stream>>>(
            h, coord, ei, pWe1, be1, pWe2, be2, pWc1, bc1, pWc2, bc2,
            w257, m_i, coord_out, E_);
    }
    // node phase
    {
        int tiles  = (N_ + 15) / 16;
        int blocks = (tiles + WAVES_PER_BLK - 1) / WAVES_PER_BLK;
        egcl_node_kernel<<<blocks, 128, 0, stream>>>(
            h, m_i, pWn1, bn1, pWn2, bn2, h_out, N_);
    }
}
